// Longformer_79791902425151
// MI455X (gfx1250) — matrix operand order repaired and checked by reading, not compiled, not run on hardware
//
#include <hip/hip_runtime.h>
#include <hip/hip_bf16.h>

// ---------------------------------------------------------------------------
// Longformer attention for MI455X (gfx1250, wave32, WMMA bf16, async-to-LDS).
// Pipeline:
//   1) gemm_proj (bf16 WMMA): q=x@Wq*scale, k=x@Wk, v=x@Wv, kg=x@Wkg, vg=x@Wvg
//   2) band_attn: flash-style banded attention (window 3W) + global column
//      - K tile staged with global_load_async_to_lds_b128 (ASYNCcnt path)
//   3) global_attn: row-0 global attention with qg = x[:,0]@Wqg
//   4) gemm_proj: out = attn @ Wo (fp32 out, bf16 A tile staged via async DMA)
// ---------------------------------------------------------------------------

#define BATCH  2
#define SEQ    4096
#define DMODEL 768
#define NHEAD  12
#define HDIM   64
#define WWIN   256
#define NCHUNK 16
#define NEGV   (-1.0e9f)

typedef __attribute__((ext_vector_type(16))) __bf16 v16bf;
typedef __attribute__((ext_vector_type(8)))  float  v8f;

union Frag {           // one WMMA 16-bit operand: 8 VGPRs = 32 bytes/lane
    v16bf v;
    uint4 q[2];
};

static __device__ inline __bf16 f2bf(float f) { return (__bf16)f; }
static __device__ inline float  bf2f(__bf16 b) { return (float)b; }

// LDS byte address of a __shared__ element (low 32 bits of the pointer)
static __device__ inline unsigned int lds_addr(const void* p) {
    return (unsigned int)(unsigned long long)p;
}
// Async DMA: 16B global -> LDS, tracked by ASYNCcnt.
static __device__ inline void async_g2l_b128(unsigned int lds, const void* g) {
    asm volatile("global_load_async_to_lds_b128 %0, %1, off"
                 :: "v"(lds), "v"(g) : "memory");
}
static __device__ inline void wait_asynccnt0() {
    asm volatile("s_wait_asynccnt 0" ::: "memory");
}

// ---------------------------------------------------------------------------
// Tiled GEMM: Y[M x D] = X[M x D] @ W[D x D], bf16 WMMA, fp32 accumulate.
// Block tile 128x64, K-step 64. 8 waves, each wave owns a 32x32 C tile.
// ---------------------------------------------------------------------------
template <bool XF32, bool OUTF32>
__global__ __launch_bounds__(256) void gemm_proj(const void* __restrict__ Xv,
                                                 const float* __restrict__ Wm,
                                                 void* __restrict__ Yv,
                                                 float scale)
{
    constexpr int BM = 128, BN = 64, BK = 64;
    constexpr int LDA = 72, LDB = 72;          // padded rows (144B): aligned + conflict-free
    __shared__ __bf16 As[BM * LDA];            // [row][k]
    __shared__ __bf16 Bs[BN * LDB];            // [n][k]  (W tile transposed)

    const int tid  = threadIdx.x;
    const int lane = tid & 31;
    const int wave = tid >> 5;
    const int half = lane >> 4, lm = lane & 15;
    const int m0 = blockIdx.y * BM;
    const int n0 = blockIdx.x * BN;
    const int wm = wave >> 1, wn = wave & 1;

    const float*  Xf = (const float*)Xv;
    const __bf16* Xb = (const __bf16*)Xv;

    v8f acc[2][2] = {};

    for (int k0 = 0; k0 < DMODEL; k0 += BK) {
        __syncthreads();
        // ---- stage A tile ----
        if (XF32) {
#pragma unroll
            for (int r = 0; r < 8; ++r) {
                int i = tid + 256 * r;                 // 2048 float4
                int row = i >> 4, c4 = (i & 15) * 4;
                const float* src = Xf + (size_t)(m0 + row) * DMODEL + k0 + c4;
                float4 f = *(const float4*)src;
                __bf16* dst = &As[row * LDA + c4];
                dst[0] = f2bf(f.x); dst[1] = f2bf(f.y);
                dst[2] = f2bf(f.z); dst[3] = f2bf(f.w);
            }
            if (k0 + BK < DMODEL)                       // prefetch next A tile
                __builtin_prefetch(Xf + (size_t)(m0 + (tid >> 1)) * DMODEL + k0 + BK, 0, 1);
        } else {
            // pure bf16 copy: async DMA straight into LDS (ASYNCcnt path)
#pragma unroll
            for (int r = 0; r < 4; ++r) {
                int i = tid + 256 * r;                 // 1024 x (8 bf16)
                int row = i >> 3, c8 = (i & 7) * 8;
                async_g2l_b128(lds_addr(&As[row * LDA + c8]),
                               (const void*)(Xb + (size_t)(m0 + row) * DMODEL + k0 + c8));
            }
        }
        // ---- stage B tile transposed: Bs[n][kk] = W[k0+kk][n0+n] ----
#pragma unroll
        for (int r = 0; r < 4; ++r) {
            int i = tid + 256 * r;                     // 1024 float4
            int kk = i >> 4, n4 = (i & 15) * 4;
            float4 f = *(const float4*)(Wm + (size_t)(k0 + kk) * DMODEL + n0 + n4);
            Bs[(n4 + 0) * LDB + kk] = f2bf(f.x);
            Bs[(n4 + 1) * LDB + kk] = f2bf(f.y);
            Bs[(n4 + 2) * LDB + kk] = f2bf(f.z);
            Bs[(n4 + 3) * LDB + kk] = f2bf(f.w);
        }
        if (XF32 && (k0 + BK < DMODEL))                 // prefetch next W tile
            __builtin_prefetch(Wm + (size_t)(k0 + BK + (tid >> 3)) * DMODEL + n0, 0, 1);
        if (!XF32) wait_asynccnt0();
        __syncthreads();

        // ---- fragments + WMMA (2 K-frags of 32 per step) ----
        Frag a[2][2], bfr[2][2];
#pragma unroll
        for (int mt = 0; mt < 2; ++mt) {
            int m = wm * 32 + mt * 16 + lm;            // A: lane = row M
#pragma unroll
            for (int kf = 0; kf < 2; ++kf) {
                a[mt][kf].q[0] = *(const uint4*)&As[m * LDA + kf * 32 + half * 8];
                a[mt][kf].q[1] = *(const uint4*)&As[m * LDA + kf * 32 + 16 + half * 8];
            }
        }
#pragma unroll
        for (int nt = 0; nt < 2; ++nt) {
            int n = wn * 32 + nt * 16 + lm;            // B: lane = col N
#pragma unroll
            for (int kf = 0; kf < 2; ++kf) {
                bfr[nt][kf].q[0] = *(const uint4*)&Bs[n * LDB + kf * 32 + half * 8];
                bfr[nt][kf].q[1] = *(const uint4*)&Bs[n * LDB + kf * 32 + 16 + half * 8];
            }
        }
#pragma unroll
        for (int mt = 0; mt < 2; ++mt)
#pragma unroll
            for (int nt = 0; nt < 2; ++nt) {
                acc[mt][nt] = __builtin_amdgcn_wmma_f32_16x16x32_bf16(
                    false, a[mt][0].v, false, bfr[nt][0].v, (short)0, acc[mt][nt], false, false);
                acc[mt][nt] = __builtin_amdgcn_wmma_f32_16x16x32_bf16(
                    false, a[mt][1].v, false, bfr[nt][1].v, (short)0, acc[mt][nt], false, false);
            }
    }

    // ---- store C (row = j + 8*half, col = lane%16) ----
#pragma unroll
    for (int mt = 0; mt < 2; ++mt)
#pragma unroll
        for (int nt = 0; nt < 2; ++nt)
#pragma unroll
            for (int j = 0; j < 8; ++j) {
                int row = m0 + wm * 32 + mt * 16 + j + 8 * half;
                int col = n0 + wn * 32 + nt * 16 + lm;
                float vo = acc[mt][nt][j] * scale;
                if constexpr (OUTF32)
                    ((float*)Yv)[(size_t)row * DMODEL + col] = vo;
                else
                    ((__bf16*)Yv)[(size_t)row * DMODEL + col] = f2bf(vo);
            }
}

// ---------------------------------------------------------------------------
// Banded attention: one block per (b, h, chunk). 8 waves x 32 query rows.
// Flash-style streaming over 24 key tiles of 32 within the 3W window, plus
// the single global key (position 0) as one extra online-softmax column.
// ---------------------------------------------------------------------------
__global__ __launch_bounds__(256) void band_attn(const __bf16* __restrict__ q,
                                                 const __bf16* __restrict__ k,
                                                 const __bf16* __restrict__ v,
                                                 const int* __restrict__ amask,
                                                 __bf16* __restrict__ out)
{
    constexpr int LDK = 72;   // 32 x 64 K tile rows padded to 144B
    constexpr int LDV = 40;   // 64 x 32 transposed V tile rows padded to 80B
    constexpr int LDP = 40;   // 16 x 32 per-wave P staging rows padded to 80B
    __shared__ __bf16 kts[32 * LDK];
    __shared__ __bf16 vts[64 * LDV];
    __shared__ __bf16 pst[8 * 16 * LDP];

    const int tid  = threadIdx.x;
    const int lane = tid & 31;
    const int wave = tid >> 5;
    const int half = lane >> 4, lm = lane & 15;
    const int c = blockIdx.x % NCHUNK;
    const int h = (blockIdx.x / NCHUNK) % NHEAD;
    const int b = blockIdx.x / (NCHUNK * NHEAD);

    const size_t bh = ((size_t)b * SEQ) * DMODEL + h * HDIM;
    const int r0 = wave * 32;                      // first query row for this wave

    // ---- Q fragments (A operand), 2 M-tiles x 2 K-frags over HD=64 ----
    Frag qf[2][2];
#pragma unroll
    for (int mt = 0; mt < 2; ++mt) {
        int srow = c * WWIN + r0 + mt * 16 + lm;
        const __bf16* qp = q + bh + (size_t)srow * DMODEL;
#pragma unroll
        for (int kf = 0; kf < 2; ++kf) {
            qf[mt][kf].q[0] = *(const uint4*)(qp + kf * 32 + half * 8);
            qf[mt][kf].q[1] = *(const uint4*)(qp + kf * 32 + 16 + half * 8);
        }
    }

    v8f  acc[2][4] = {};
    float m_run[2][8], l_run[2][8];
#pragma unroll
    for (int mt = 0; mt < 2; ++mt)
#pragma unroll
        for (int j = 0; j < 8; ++j) { m_run[mt][j] = -__builtin_inff(); l_run[mt][j] = 0.f; }

    for (int kt = 0; kt < 24; ++kt) {
        __syncthreads();
        // ---- staging: K via async global->LDS DMA, V via register transpose ----
        {
            int key = tid >> 3, c8 = (tid & 7) * 8;
            int kj = kt * 32 + key;                    // 0..767 within window
            int kp = c * WWIN + kj - WWIN;             // absolute key position
            if (kp >= 0 && kp < SEQ) {
                async_g2l_b128(lds_addr(&kts[key * LDK + c8]),
                               (const void*)(k + bh + (size_t)kp * DMODEL + c8));
                uint4 vd = *(const uint4*)(v + bh + (size_t)kp * DMODEL + c8);
                const __bf16* vh = (const __bf16*)&vd;
#pragma unroll
                for (int e = 0; e < 8; ++e) vts[(c8 + e) * LDV + key] = vh[e];
            } else {
                uint4 z = {0, 0, 0, 0};
                *(uint4*)&kts[key * LDK + c8] = z;
                __bf16 zb = f2bf(0.f);
#pragma unroll
                for (int e = 0; e < 8; ++e) vts[(c8 + e) * LDV + key] = zb;
            }
        }
        wait_asynccnt0();
        __syncthreads();

        // ---- B fragments: V (P@V) and K (Q@K^T) ----
        Frag vf[4];
#pragma unroll
        for (int nh = 0; nh < 4; ++nh) {
            int n = nh * 16 + lm;                      // n = head-dim column
            vf[nh].q[0] = *(const uint4*)&vts[n * LDV + half * 8];
            vf[nh].q[1] = *(const uint4*)&vts[n * LDV + 16 + half * 8];
        }
        Frag kb[2][2];
#pragma unroll
        for (int nt = 0; nt < 2; ++nt) {
            int n = nt * 16 + lm;                      // n = key within tile
#pragma unroll
            for (int kf = 0; kf < 2; ++kf) {
                kb[nt][kf].q[0] = *(const uint4*)&kts[n * LDK + kf * 32 + half * 8];
                kb[nt][kf].q[1] = *(const uint4*)&kts[n * LDK + kf * 32 + 16 + half * 8];
            }
        }

#pragma unroll
        for (int mt = 0; mt < 2; ++mt) {
            // scores: 16 rows x 32 keys
            v8f sacc[2] = {};
#pragma unroll
            for (int nt = 0; nt < 2; ++nt) {
                sacc[nt] = __builtin_amdgcn_wmma_f32_16x16x32_bf16(
                    false, qf[mt][0].v, false, kb[nt][0].v, (short)0, sacc[nt], false, false);
                sacc[nt] = __builtin_amdgcn_wmma_f32_16x16x32_bf16(
                    false, qf[mt][1].v, false, kb[nt][1].v, (short)0, sacc[nt], false, false);
            }

            // mask + online softmax (row = j + 8*half within the M tile)
            float p[2][8], rmax[8];
            const int qi_base = r0 + mt * 16;
#pragma unroll
            for (int j = 0; j < 8; ++j) {
                int qi = qi_base + j + 8 * half;
                float best = NEGV;
#pragma unroll
                for (int nt = 0; nt < 2; ++nt) {
                    int kj = kt * 32 + nt * 16 + lm;
                    int kp = c * WWIN + kj - WWIN;
                    bool ok = (kj >= qi) && (kj <= qi + 2 * WWIN) &&
                              (kp >= 0) && (kp < SEQ) && (kp != 0);
                    float s = sacc[nt][j];
                    s = ok ? (s + ((amask[b * SEQ + (ok ? kp : 0)] > 0) ? 0.f : NEGV)) : NEGV;
                    p[nt][j] = s;
                    best = fmaxf(best, s);
                }
                rmax[j] = best;
            }
#pragma unroll
            for (int j = 0; j < 8; ++j) {
#pragma unroll
                for (int off = 1; off < 16; off <<= 1)
                    rmax[j] = fmaxf(rmax[j], __shfl_xor(rmax[j], off, 32));
                float mn = fmaxf(m_run[mt][j], rmax[j]);
                float sf = __expf(m_run[mt][j] - mn);
                m_run[mt][j] = mn;
                float rs = 0.f;
#pragma unroll
                for (int nt = 0; nt < 2; ++nt) {
                    p[nt][j] = __expf(p[nt][j] - mn);
                    rs += p[nt][j];
                }
#pragma unroll
                for (int off = 1; off < 16; off <<= 1)
                    rs += __shfl_xor(rs, off, 32);
                l_run[mt][j] = l_run[mt][j] * sf + rs;
#pragma unroll
                for (int nh = 0; nh < 4; ++nh) acc[mt][nh][j] *= sf;
            }

            // re-lay P (C layout) into A layout via per-wave LDS patch
            __bf16* pw = &pst[(wave * 16) * LDP];
#pragma unroll
            for (int nt = 0; nt < 2; ++nt)
#pragma unroll
                for (int j = 0; j < 8; ++j)
                    pw[(j + 8 * half) * LDP + nt * 16 + lm] = f2bf(p[nt][j]);
            asm volatile("s_wait_dscnt 0" ::: "memory");
            Frag pa;
            pa.q[0] = *(const uint4*)&pw[lm * LDP + half * 8];
            pa.q[1] = *(const uint4*)&pw[lm * LDP + 16 + half * 8];

            // acc += P (16x32) @ V (32x64)
#pragma unroll
            for (int nh = 0; nh < 4; ++nh)
                acc[mt][nh] = __builtin_amdgcn_wmma_f32_16x16x32_bf16(
                    false, pa.v, false, vf[nh].v, (short)0, acc[mt][nh], false, false);
        }
    }

    // ---- global key (absolute position 0) as one extra column ----
    const __bf16* k0p = k + bh;
    const __bf16* v0p = v + bh;
    const float madd0 = (amask[b * SEQ] > 0) ? 0.f : NEGV;
    float v0d[4], k0d[4];
#pragma unroll
    for (int nh = 0; nh < 4; ++nh) v0d[nh] = bf2f(v0p[nh * 16 + lm]);
#pragma unroll
    for (int e = 0; e < 4; ++e) k0d[e] = bf2f(k0p[lm * 4 + e]);

#pragma unroll
    for (int mt = 0; mt < 2; ++mt)
#pragma unroll
        for (int j = 0; j < 8; ++j) {
            int srow = c * WWIN + r0 + mt * 16 + j + 8 * half;
            const __bf16* qp = q + bh + (size_t)srow * DMODEL;
            float g = 0.f;
#pragma unroll
            for (int e = 0; e < 4; ++e) g += bf2f(qp[lm * 4 + e]) * k0d[e];
#pragma unroll
            for (int off = 1; off < 16; off <<= 1) g += __shfl_xor(g, off, 32);
            g += madd0;
            float mn = fmaxf(m_run[mt][j], g);
            float sf = __expf(m_run[mt][j] - mn);
            float pg = __expf(g - mn);
            l_run[mt][j] = l_run[mt][j] * sf + pg;
            m_run[mt][j] = mn;
#pragma unroll
            for (int nh = 0; nh < 4; ++nh)
                acc[mt][nh][j] = acc[mt][nh][j] * sf + pg * v0d[nh];
        }

    // ---- finalize & store bf16 in (b, s, h*HD + hd) layout ----
#pragma unroll
    for (int mt = 0; mt < 2; ++mt)
#pragma unroll
        for (int j = 0; j < 8; ++j) {
            float inv = 1.f / l_run[mt][j];
            int srow = c * WWIN + r0 + mt * 16 + j + 8 * half;
            __bf16* op = out + bh + (size_t)srow * DMODEL;
#pragma unroll
            for (int nh = 0; nh < 4; ++nh)
                op[nh * 16 + lm] = f2bf(acc[mt][nh][j] * inv);
        }
}

// ---------------------------------------------------------------------------
// Global row (s==0) attention: one block per (b,h). qg0 computed directly.
// ---------------------------------------------------------------------------
__global__ __launch_bounds__(256) void global_attn(const float* __restrict__ x,
                                                   const float* __restrict__ Wqg,
                                                   const __bf16* __restrict__ kg,
                                                   const __bf16* __restrict__ vg,
                                                   const int* __restrict__ amask,
                                                   __bf16* __restrict__ out)
{
    __shared__ float sc[SEQ];
    __shared__ float qg_s[HDIM];
    __shared__ float part[4][HDIM];
    __shared__ float red[256];

    const int tid = threadIdx.x;
    const int h = blockIdx.x % NHEAD;
    const int b = blockIdx.x / NHEAD;
    const float scale = 0.125f;                     // 1/sqrt(64)
    const size_t bh = ((size_t)b * SEQ) * DMODEL + h * HDIM;

    // qg0[d] = scale * sum_kk x[b,0,kk] * Wqg[kk, h*HD+d]
    {
        int d = tid & 63, pr = tid >> 6;
        float a = 0.f;
        for (int kk = pr * 192; kk < (pr + 1) * 192; ++kk)
            a += x[(size_t)b * SEQ * DMODEL + kk] * Wqg[(size_t)kk * DMODEL + h * HDIM + d];
        part[pr][d] = a;
    }
    __syncthreads();
    if (tid < HDIM) qg_s[tid] = (part[0][tid] + part[1][tid] + part[2][tid] + part[3][tid]) * scale;
    __syncthreads();

    // scores over all S keys
    float lmax = -__builtin_inff();
    for (int s = tid; s < SEQ; s += 256) {
        const __bf16* kp = kg + bh + (size_t)s * DMODEL;
        float g = 0.f;
#pragma unroll
        for (int d = 0; d < HDIM; ++d) g += qg_s[d] * bf2f(kp[d]);
        g += (amask[b * SEQ + s] > 0) ? 0.f : NEGV;
        sc[s] = g;
        lmax = fmaxf(lmax, g);
    }
    red[tid] = lmax;
    __syncthreads();
    for (int st = 128; st > 0; st >>= 1) {
        if (tid < st) red[tid] = fmaxf(red[tid], red[tid + st]);
        __syncthreads();
    }
    float gmax = red[0];
    __syncthreads();
    float lsum = 0.f;
    for (int s = tid; s < SEQ; s += 256) {
        float p = __expf(sc[s] - gmax);
        sc[s] = p;
        lsum += p;
    }
    red[tid] = lsum;
    __syncthreads();
    for (int st = 128; st > 0; st >>= 1) {
        if (tid < st) red[tid] += red[tid + st];
        __syncthreads();
    }
    float inv = 1.f / red[0];
    __syncthreads();

    // weighted sum of vg
    {
        int d = tid & 63, pr = tid >> 6;
        float a = 0.f;
        for (int s = pr * 1024; s < (pr + 1) * 1024; ++s)
            a += sc[s] * bf2f(vg[bh + (size_t)s * DMODEL + d]);
        part[pr][d] = a;
    }
    __syncthreads();
    if (tid < HDIM) {
        float og = (part[0][tid] + part[1][tid] + part[2][tid] + part[3][tid]) * inv;
        out[bh + tid] = f2bf(og);   // overwrite band output at s == 0
    }
}

// ---------------------------------------------------------------------------
extern "C" void kernel_launch(void* const* d_in, const int* in_sizes, int n_in,
                              void* d_out, int out_size, void* d_ws, size_t ws_size,
                              hipStream_t stream)
{
    const float* x   = (const float*)d_in[0];
    const float* Wq  = (const float*)d_in[1];
    const float* Wk  = (const float*)d_in[2];
    const float* Wv  = (const float*)d_in[3];
    const float* Wqg = (const float*)d_in[4];
    const float* Wkg = (const float*)d_in[5];
    const float* Wvg = (const float*)d_in[6];
    const float* Wo  = (const float*)d_in[7];
    const int*   am  = (const int*)d_in[8];

    // workspace: 6 bf16 buffers of B*S*D (q, k, v, kg, vg, attn) ~ 75.5 MB
    const size_t NB = (size_t)BATCH * SEQ * DMODEL;
    __bf16* qb  = (__bf16*)d_ws;
    __bf16* kb  = qb + NB;
    __bf16* vb  = kb + NB;
    __bf16* kgb = vb + NB;
    __bf16* vgb = kgb + NB;
    __bf16* ab  = vgb + NB;

    dim3 gg(DMODEL / 64, (BATCH * SEQ) / 128);
    gemm_proj<true, false><<<gg, 256, 0, stream>>>(x, Wq,  qb,  0.125f);
    gemm_proj<true, false><<<gg, 256, 0, stream>>>(x, Wk,  kb,  1.0f);
    gemm_proj<true, false><<<gg, 256, 0, stream>>>(x, Wv,  vb,  1.0f);
    gemm_proj<true, false><<<gg, 256, 0, stream>>>(x, Wkg, kgb, 1.0f);
    gemm_proj<true, false><<<gg, 256, 0, stream>>>(x, Wvg, vgb, 1.0f);

    band_attn<<<BATCH * NHEAD * NCHUNK, 256, 0, stream>>>(qb, kb, vb, am, ab);
    global_attn<<<BATCH * NHEAD, 256, 0, stream>>>(x, Wqg, kgb, vgb, am, ab);

    gemm_proj<false, true><<<gg, 256, 0, stream>>>(ab, Wo, d_out, 1.0f);
}
